// Quantize_55284819034574
// MI455X (gfx1250) — matrix operand order repaired and checked by reading, not compiled, not run on hardware
//
#include <hip/hip_runtime.h>
#include <hip/hip_bf16.h>

// ---------------------------------------------------------------------------
// VQ codebook lookup (vector quantization) for MI455X / gfx1250.
//   input [8,2048,512] f32, embed [512,8192] f32
//   out = (quantize_st [8,2048,512], loss scalar, embed_ind [8,2048]) flat f32
// Distance GEMM (16384x8192x512, 137 GFLOP) runs on v_wmma_f32_16x16x32_bf16
// with argmin fused (dist matrix never materialized; 512 MB avoided).
// Inner loop: 2 M-tiles x 2 N-tiles per K-step (4 WMMA per step), B fragments
// prefetched 2 K-steps ahead (3-buffer rotation), A re-read from LDS each
// iteration. An iteration-scoped compiler barrier stops LICM from hoisting
// the 256-VGPR A set out of the nt loop (which caused MSB-mode/spills).
// Outputs use exact f32 codebook values (gather from transposed f32 copy).
// ---------------------------------------------------------------------------

typedef __attribute__((ext_vector_type(16))) __bf16 v16bf;
typedef __attribute__((ext_vector_type(8)))  __bf16 v8bf;
typedef __attribute__((ext_vector_type(8)))  float  v8f;

#define DIM   512
#define NEMB  8192
#define NTOK  16384      // 8 * 2048
#define MT    32         // tokens per block in the argmin kernel
#define LDSS  (DIM + 8)  // LDS row stride (bf16 elems); 520*2 = 1040 B (16B multiple)

// B fragment (32x16 bf16): lane = col, K = ks*32 + half*16 + {0..15}
#define LOAD_B(dst, base, ks) do {                       \
    const __bf16* _p = (base) + (ks) * 32 + half * 16;   \
    ((v8bf*)&(dst))[0] = *(const v8bf*)(_p);             \
    ((v8bf*)&(dst))[1] = *(const v8bf*)(_p + 8);         \
  } while (0)

// A fragment (16x32 bf16): lane = row, K = ks*32 + half*16 + {0..15}
#define LOAD_A(dst, arow, ks) do {                       \
    const __bf16* _p = (arow) + (ks) * 32 + half * 16;   \
    ((v8bf*)&(dst))[0] = *(const v8bf*)(_p);             \
    ((v8bf*)&(dst))[1] = *(const v8bf*)(_p + 8);         \
  } while (0)

#define WMMA_BF16(a, b, c) \
  __builtin_amdgcn_wmma_f32_16x16x32_bf16(false, (a), false, (b), (short)0, (c), false, false)

// --- kernel 1: embed [DIM][NEMB] -> Ebt bf16 [NEMB][DIM] + Etf f32 [NEMB][DIM]
__global__ __launch_bounds__(256) void vq_prep_embed(
    const float* __restrict__ E, __bf16* __restrict__ Ebt, float* __restrict__ Etf) {
  __shared__ float tile[32][33];
  const int tx = threadIdx.x, ty = threadIdx.y;          // block (32, 8)
  const int e0 = blockIdx.x * 32, c0 = blockIdx.y * 32;
#pragma unroll
  for (int j = 0; j < 4; ++j)
    tile[ty + j * 8][tx] = E[(size_t)(c0 + ty + j * 8) * NEMB + e0 + tx];
  __syncthreads();
#pragma unroll
  for (int j = 0; j < 4; ++j) {
    float v = tile[tx][ty + j * 8];
    const size_t o = (size_t)(e0 + ty + j * 8) * DIM + c0 + tx;
    Etf[o] = v;
    Ebt[o] = (__bf16)v;
  }
}

// --- kernel 2: codenorm[e] = sum_c E[c][e]^2 ; also zero the loss accumulator
__global__ __launch_bounds__(256) void vq_codenorm(
    const float* __restrict__ E, float* __restrict__ codenorm, float* __restrict__ lossAcc) {
  const int e = blockIdx.x * 256 + threadIdx.x;
  float s = 0.f;
  for (int c = 0; c < DIM; ++c) {
    float v = E[(size_t)c * NEMB + e];
    s = fmaf(v, v, s);
  }
  codenorm[e] = s;
  if (e == 0) *lossAcc = 0.f;
}

// --- kernel 3: fused bf16 WMMA distance GEMM + argmin
// block = 256 threads (8 waves), owns MT=32 tokens (2 M-tiles); wave w scans
// codes [w*1024, (w+1)*1024) two 16-wide N tiles at a time.
__global__ __launch_bounds__(256) void vq_argmin_kernel(
    const float* __restrict__ X,        // [NTOK][DIM] f32
    const __bf16* __restrict__ Ebt,     // [NEMB][DIM] bf16 (K contiguous)
    const float* __restrict__ codenorm, // [NEMB]
    float* __restrict__ outInd) {       // [NTOK], float-coded index
  __shared__ __bf16 Xs[MT * LDSS];
  __shared__ float redV[8][MT];
  __shared__ int   redI[8][MT];

  const int tid  = threadIdx.x;
  const int wave = tid >> 5;
  const int lane = tid & 31;
  const int half = lane >> 4;
  const int l15  = lane & 15;
  const int m0   = blockIdx.x * MT;

  // Stage X tile into LDS as bf16 (f32 -> bf16 conversion on the fly).
  for (int i = tid; i < MT * DIM / 2; i += 256) {
    const int row = i / (DIM / 2);
    const int c2  = i % (DIM / 2);
    float2 f = ((const float2*)(X + (size_t)(m0 + row) * DIM))[c2];
    Xs[row * LDSS + c2 * 2]     = (__bf16)f.x;
    Xs[row * LDSS + c2 * 2 + 1] = (__bf16)f.y;
  }
  __syncthreads();

  float minv[2][8];
  int   mini[2][8];
#pragma unroll
  for (int t = 0; t < 2; ++t)
#pragma unroll
    for (int r = 0; r < 8; ++r) { minv[t][r] = 3.4e38f; mini[t][r] = 0; }

  const int n0w = wave * (NEMB / 8);                 // 1024 codes per wave
  const __bf16* arow0 = Xs + (0 * 16 + l15) * LDSS;  // A row for M-tile 0
  const __bf16* arow1 = Xs + (1 * 16 + l15) * LDSS;  // A row for M-tile 1

  for (int nt = 0; nt < (NEMB / 8) / 32; ++nt) {     // 32 iterations x 2 N tiles
    // Stop LICM from hoisting the (nt-invariant) A-fragment LDS loads out of
    // this loop: hoisting all 16 K-steps of A costs 256 VGPRs and previously
    // forced VGPR-MSB mode / scratch spills. A stays a cheap per-iteration
    // ds_load_b128 stream instead.
    asm volatile("" ::: "memory");

    const int nA = n0w + nt * 32 + l15;              // lower-index column set
    const int nB = nA + 16;                          // higher-index column set
    const __bf16* browA = Ebt + (size_t)nA * DIM;
    const __bf16* browB = Ebt + (size_t)nB * DIM;

    v8f accA0, accA1, accB0, accB1;                  // [Ntile][Mtile]
#pragma unroll
    for (int r = 0; r < 8; ++r) { accA0[r] = 0.f; accA1[r] = 0.f; accB0[r] = 0.f; accB1[r] = 0.f; }

    // B prefetched 2 K-steps ahead (3-buffer rotation); A 1 step ahead from LDS.
    v16bf bA[3], bB[3], a0[2], a1[2];
    LOAD_B(bA[0], browA, 0);
    LOAD_B(bB[0], browB, 0);
    LOAD_B(bA[1], browA, 1);
    LOAD_B(bB[1], browB, 1);
    LOAD_A(a0[0], arow0, 0);
    LOAD_A(a1[0], arow1, 0);
#pragma unroll
    for (int ks = 0; ks < 16; ++ks) {
      const int cur  = ks % 3;
      const int acur = ks & 1, anxt = acur ^ 1;
      if (ks < 15) {
        LOAD_A(a0[anxt], arow0, ks + 1);
        LOAD_A(a1[anxt], arow1, ks + 1);
      }
      if (ks < 14) {
        const int pre = (ks + 2) % 3;
        LOAD_B(bA[pre], browA, ks + 2);
        LOAD_B(bB[pre], browB, ks + 2);
      }
      accA0 = WMMA_BF16(a0[acur], bA[cur], accA0);
      accA1 = WMMA_BF16(a1[acur], bA[cur], accA1);
      accB0 = WMMA_BF16(a0[acur], bB[cur], accB0);
      accB1 = WMMA_BF16(a1[acur], bB[cur], accB1);
    }

    // dist (up to +||x||^2 row constant) = ||e||^2 - 2 x.e ; evaluate the
    // lower-index tile first with strict < to keep argmin's first-match rule.
    const float cnA = codenorm[nA];
    const float cnB = codenorm[nB];
#pragma unroll
    for (int r = 0; r < 8; ++r) {
      float d = fmaf(-2.f, accA0[r], cnA);
      if (d < minv[0][r]) { minv[0][r] = d; mini[0][r] = nA; }
      d = fmaf(-2.f, accB0[r], cnB);
      if (d < minv[0][r]) { minv[0][r] = d; mini[0][r] = nB; }
      d = fmaf(-2.f, accA1[r], cnA);
      if (d < minv[1][r]) { minv[1][r] = d; mini[1][r] = nA; }
      d = fmaf(-2.f, accB1[r], cnB);
      if (d < minv[1][r]) { minv[1][r] = d; mini[1][r] = nB; }
    }
  }

  // Reduce across the 16-lane column group (ties -> lowest index, like argmin).
#pragma unroll
  for (int t = 0; t < 2; ++t)
#pragma unroll
    for (int r = 0; r < 8; ++r) {
      float v = minv[t][r];
      int   i = mini[t][r];
#pragma unroll
      for (int off = 8; off >= 1; off >>= 1) {
        const float ov = __shfl_xor(v, off, 32);
        const int   oi = __shfl_xor(i, off, 32);
        if (ov < v || (ov == v && oi < i)) { v = ov; i = oi; }
      }
      if (l15 == 0) {  // C layout: lanes 0-15 hold row r, lanes 16-31 row r+8
        const int row = t * 16 + half * 8 + r;
        redV[wave][row] = v;
        redI[wave][row] = i;
      }
    }
  __syncthreads();

  // Reduce across the 8 waves (disjoint code slices).
  if (tid < MT) {
    float bv = redV[0][tid];
    int   bi = redI[0][tid];
#pragma unroll
    for (int w = 1; w < 8; ++w) {
      const float v = redV[w][tid];
      const int   i = redI[w][tid];
      if (v < bv || (v == bv && i < bi)) { bv = v; bi = i; }
    }
    outInd[m0 + tid] = (float)bi;
  }
}

// --- kernel 4: gather exact f32 codebook row, straight-through output, loss sum
__global__ __launch_bounds__(256) void vq_gather_loss(
    const float* __restrict__ X, const float* __restrict__ Etf,
    const float* __restrict__ indF, float* __restrict__ outQ,
    float* __restrict__ lossAcc) {
  const int tid  = threadIdx.x;
  const int wave = tid >> 5, lane = tid & 31;
  const int tok  = blockIdx.x * 8 + wave;
  const int idx  = (int)indF[tok];
  const float* e = Etf + (size_t)idx * DIM;
  const float* x = X + (size_t)tok * DIM;
  float* q = outQ + (size_t)tok * DIM;
  float s = 0.f;
  for (int c = lane; c < DIM; c += 32) {
    const float qv = e[c];
    const float xv = x[c];
    const float d  = qv - xv;   // stop_grad(q - x)
    q[c] = xv + d;              // straight-through estimator, reference rounding
    s = fmaf(d, d, s);
  }
#pragma unroll
  for (int off = 16; off >= 1; off >>= 1) s += __shfl_xor(s, off, 32);
  if (lane == 0) atomicAdd(lossAcc, s);
}

// --- kernel 5: finalize loss scalar
__global__ void vq_finalize_loss(const float* __restrict__ lossAcc,
                                 float* __restrict__ outLoss) {
  *outLoss = *lossAcc * (0.25f / (float)(NTOK * DIM));
}

extern "C" void kernel_launch(void* const* d_in, const int* in_sizes, int n_in,
                              void* d_out, int out_size, void* d_ws, size_t ws_size,
                              hipStream_t stream) {
  const float* X = (const float*)d_in[0];   // [8,2048,512]
  const float* E = (const float*)d_in[1];   // [512,8192]
  float* out     = (float*)d_out;
  float* outQ    = out;                      // 8388608 floats
  float* outLoss = out + (size_t)NTOK * DIM; // 1 float
  float* outInd  = outLoss + 1;              // 16384 floats (float-coded int)

  char* ws = (char*)d_ws;
  __bf16* Ebt     = (__bf16*)ws;                                   // 8 MB
  float*  Etf     = (float*)(ws + (size_t)NEMB * DIM * 2);         // 16 MB
  float*  cnorm   = (float*)(ws + (size_t)NEMB * DIM * 6);         // 32 KB
  float*  lossAcc = (float*)(ws + (size_t)NEMB * DIM * 6 + NEMB * 4);

  vq_prep_embed   <<<dim3(NEMB / 32, DIM / 32), dim3(32, 8), 0, stream>>>(E, Ebt, Etf);
  vq_codenorm     <<<NEMB / 256, 256, 0, stream>>>(E, cnorm, lossAcc);
  vq_argmin_kernel<<<NTOK / MT, 256, 0, stream>>>(X, Ebt, cnorm, outInd);
  vq_gather_loss  <<<NTOK / 8, 256, 0, stream>>>(X, Etf, outInd, outQ, lossAcc);
  vq_finalize_loss<<<1, 1, 0, stream>>>(lossAcc, outLoss);
}
